// FP8StaticLinearQuantizer_13477607375255
// MI455X (gfx1250) — 8-bit matrix operand order repaired and checked by reading, compiled, not run on hardware
//
#include <hip/hip_runtime.h>

typedef __attribute__((ext_vector_type(16))) int   v16i;
typedef __attribute__((ext_vector_type(8)))  float v8f;

#define FP8_MAX 448.0f

__device__ __forceinline__ unsigned int f32_to_e4m3(float f) {
  unsigned int sign = (__float_as_uint(f) >> 24) & 0x80u;
  float a = fabsf(f);
  if (!(a <= FP8_MAX)) a = FP8_MAX;
  if (a == 0.0f) return sign;
  int unb = (int)(__float_as_uint(a) >> 23) - 127;
  if (unb < -6) {
    float q = rintf(a * 512.0f);
    return sign | (unsigned int)q;
  }
  float q = rintf(a * exp2f((float)(3 - unb)));
  if (q >= 16.0f) { q = 8.0f; unb += 1; }
  return sign | ((unsigned int)(unb + 7) << 3) | ((unsigned int)q & 7u);
}

__device__ __forceinline__ void async_load_b128(unsigned lds_off,
                                                unsigned long long gaddr) {
  asm volatile("global_load_async_to_lds_b128 %0, %1, off"
               :: "v"(lds_off), "v"(gaddr) : "memory");
}
__device__ __forceinline__ void wait_asynccnt0() {
  asm volatile("s_wait_asynccnt 0x0" ::: "memory");
}

__global__ void init_ws_kernel(unsigned int* a, unsigned int* b) {
  if (threadIdx.x == 0) { *a = 0u; *b = 0u; }
}

__global__ __launch_bounds__(256)
void amax_abs_kernel(const float4* __restrict__ in, long n4,
                     unsigned int* __restrict__ slot) {
  __shared__ float red[256];
  float m = 0.0f;
  for (long i = (long)blockIdx.x * blockDim.x + threadIdx.x; i < n4;
       i += (long)gridDim.x * blockDim.x) {
    float4 v = in[i];
    m = fmaxf(m, fmaxf(fmaxf(fabsf(v.x), fabsf(v.y)),
                       fmaxf(fabsf(v.z), fabsf(v.w))));
  }
  red[threadIdx.x] = m;
  __syncthreads();
  for (int s = 128; s > 0; s >>= 1) {
    if ((int)threadIdx.x < s)
      red[threadIdx.x] = fmaxf(red[threadIdx.x], red[threadIdx.x + s]);
    __syncthreads();
  }
  if (threadIdx.x == 0) atomicMax(slot, __float_as_uint(red[0]));
}

__global__ __launch_bounds__(256)
void quant_x_kernel(const float4* __restrict__ x, unsigned int* __restrict__ qx,
                    const unsigned int* __restrict__ amax_slot, long n4) {
  long i = (long)blockIdx.x * blockDim.x + threadIdx.x;
  if (i >= n4) return;
  float scale = fmaxf(__uint_as_float(*amax_slot), 1e-12f) / FP8_MAX;
  float4 v = x[i];
  unsigned int p =  f32_to_e4m3(fminf(fmaxf(v.x / scale, -FP8_MAX), FP8_MAX))
                 | (f32_to_e4m3(fminf(fmaxf(v.y / scale, -FP8_MAX), FP8_MAX)) << 8)
                 | (f32_to_e4m3(fminf(fmaxf(v.z / scale, -FP8_MAX), FP8_MAX)) << 16)
                 | (f32_to_e4m3(fminf(fmaxf(v.w / scale, -FP8_MAX), FP8_MAX)) << 24);
  qx[i] = p;
}

__global__ __launch_bounds__(256)
void quant_w_kernel(const float4* __restrict__ w, unsigned int* __restrict__ qw,
                    long n4) {
  long i = (long)blockIdx.x * blockDim.x + threadIdx.x;
  if (i >= n4) return;
  float4 v = w[i];
  qw[i] =  f32_to_e4m3(v.x)
        | (f32_to_e4m3(v.y) << 8)
        | (f32_to_e4m3(v.z) << 16)
        | (f32_to_e4m3(v.w) << 24);
}

#define TILE_STRIDE 144

__global__ __launch_bounds__(256)
void gemm_fp8_kernel(const unsigned char* __restrict__ qx,
                     const unsigned char* __restrict__ qw,
                     const unsigned int* __restrict__ amax_slot,
                     const float* __restrict__ wscale_p,
                     const float* __restrict__ bias,
                     float* __restrict__ out,
                     int M, int N, int K) {
  __shared__ __align__(16) unsigned char smA[128 * TILE_STRIDE];
  __shared__ __align__(16) unsigned char smB[128 * TILE_STRIDE];

  const int tid  = threadIdx.x;
  const int wave = tid >> 5;
  const int lane = tid & 31;
  const int t    = lane & 15;
  const int h    = lane >> 4;
  const int wm   = wave & 1;
  const int wn   = wave >> 1;
  const long arow0 = (long)blockIdx.x * 128;
  const long brow0 = (long)blockIdx.y * 128;

  const unsigned smA_base = (unsigned)(uintptr_t)&smA[0];
  const unsigned smB_base = (unsigned)(uintptr_t)&smB[0];

  v8f acc[4][2];
#pragma unroll
  for (int i = 0; i < 4; ++i)
#pragma unroll
    for (int j = 0; j < 2; ++j)
#pragma unroll
      for (int r = 0; r < 8; ++r) acc[i][j][r] = 0.0f;

  for (int k = 0; k < K; k += 128) {
    __syncthreads();
#pragma unroll
    for (int j = 0; j < 4; ++j) {
      const int c   = tid + 256 * j;
      const int row = c >> 3;
      const int o   = (c & 7) * 16;
      async_load_b128(smA_base + row * TILE_STRIDE + o,
          (unsigned long long)(uintptr_t)(qx + (arow0 + row) * (long)K + k + o));
      async_load_b128(smB_base + row * TILE_STRIDE + o,
          (unsigned long long)(uintptr_t)(qw + (brow0 + row) * (long)K + k + o));
    }
    wait_asynccnt0();
    __syncthreads();

    v16i bfrag[2];
#pragma unroll
    for (int nt = 0; nt < 2; ++nt) {
      const unsigned char* bp = smB + (wn * 32 + nt * 16 + t) * TILE_STRIDE + 16 * h;
#pragma unroll
      for (int c = 0; c < 4; ++c) {
        int4 d = *(const int4*)(bp + 32 * c);
        bfrag[nt][4 * c + 0] = d.x;
        bfrag[nt][4 * c + 1] = d.y;
        bfrag[nt][4 * c + 2] = d.z;
        bfrag[nt][4 * c + 3] = d.w;
      }
    }
#pragma unroll
    for (int mt = 0; mt < 4; ++mt) {
      const unsigned char* ap = smA + (wm * 64 + mt * 16 + t) * TILE_STRIDE + 16 * h;
      v16i afrag;
#pragma unroll
      for (int j = 0; j < 8; ++j) {
        unsigned long long d = *(const unsigned long long*)(ap + 32 * (j >> 1) + 8 * (j & 1));
        afrag[2 * j + 0] = (int)(unsigned int)(d & 0xffffffffull);
        afrag[2 * j + 1] = (int)(unsigned int)(d >> 32);
      }
      acc[mt][0] = __builtin_amdgcn_wmma_f32_16x16x128_fp8_fp8(
          afrag, bfrag[0], (short)0, acc[mt][0], false, false);
      acc[mt][1] = __builtin_amdgcn_wmma_f32_16x16x128_fp8_fp8(
          afrag, bfrag[1], (short)0, acc[mt][1], false, false);
    }
  }

  const float s = (fmaxf(__uint_as_float(*amax_slot), 1e-12f) / FP8_MAX) * (*wscale_p);
#pragma unroll
  for (int mt = 0; mt < 4; ++mt)
#pragma unroll
    for (int nt = 0; nt < 2; ++nt) {
      const long n = brow0 + wn * 32 + nt * 16 + t;
      const float bb = bias[n];
#pragma unroll
      for (int r = 0; r < 8; ++r) {
        const long m = arow0 + wm * 64 + mt * 16 + r + 8 * h;
        out[m * (long)N + n] = acc[mt][nt][r] * s + bb;
      }
    }
}

__global__ __launch_bounds__(256)
void rescale_kernel(float4* __restrict__ out,
                    const unsigned int* __restrict__ amax_slot, long n4) {
  long i = (long)blockIdx.x * blockDim.x + threadIdx.x;
  if (i >= n4) return;
  float s = fmaxf(__uint_as_float(*amax_slot), 1e-12f) / FP8_MAX;
  float4 v = out[i];
  v.x = fminf(fmaxf(v.x / s, -FP8_MAX), FP8_MAX) * s;
  v.y = fminf(fmaxf(v.y / s, -FP8_MAX), FP8_MAX) * s;
  v.z = fminf(fmaxf(v.z / s, -FP8_MAX), FP8_MAX) * s;
  v.w = fminf(fmaxf(v.w / s, -FP8_MAX), FP8_MAX) * s;
  out[i] = v;
}

extern "C" void kernel_launch(void* const* d_in, const int* in_sizes, int n_in,
                              void* d_out, int out_size, void* d_ws, size_t ws_size,
                              hipStream_t stream) {
  const float* x       = (const float*)d_in[0];
  const float* qweight = (const float*)d_in[1];
  const float* wscale  = (const float*)d_in[2];
  const float* bias    = (const float*)d_in[3];
  float* out = (float*)d_out;

  const long N = in_sizes[3];
  const long K = (long)in_sizes[1] / N;
  const long M = (long)in_sizes[0] / K;

  unsigned char* ws = (unsigned char*)d_ws;
  unsigned int* amax_x = (unsigned int*)ws;
  unsigned int* amax_o = (unsigned int*)(ws + 4);
  unsigned char* qx = ws + 256;
  unsigned char* qw = qx + (size_t)M * K;

  init_ws_kernel<<<1, 64, 0, stream>>>(amax_x, amax_o);

  const long nx4 = M * K / 4;
  const long nw4 = N * K / 4;
  const long no4 = M * N / 4;

  amax_abs_kernel<<<2048, 256, 0, stream>>>((const float4*)x, nx4, amax_x);
  quant_x_kernel<<<(unsigned)((nx4 + 255) / 256), 256, 0, stream>>>(
      (const float4*)x, (unsigned int*)qx, amax_x, nx4);
  quant_w_kernel<<<(unsigned)((nw4 + 255) / 256), 256, 0, stream>>>(
      (const float4*)qweight, (unsigned int*)qw, nw4);

  dim3 grid((unsigned)(M / 128), (unsigned)(N / 128));
  gemm_fp8_kernel<<<grid, 256, 0, stream>>>(qx, qw, amax_x, wscale, bias, out,
                                            (int)M, (int)N, (int)K);

  amax_abs_kernel<<<2048, 256, 0, stream>>>((const float4*)out, no4, amax_o);
  rescale_kernel<<<(unsigned)((no4 + 255) / 256), 256, 0, stream>>>(
      (float4*)out, amax_o, no4);
}
